// Transformer_7310034337850
// MI455X (gfx1250) — matrix operand order repaired and checked by reading, not compiled, not run on hardware
//
#include <hip/hip_runtime.h>
#include <hip/hip_bf16.h>
#include <stdint.h>

#define B_   4
#define N_   4096
#define C_   512
#define CI_  256
#define M_   (B_*N_)          // 16384 rows
#define EPSV 1e-5f

#ifndef __has_builtin
#define __has_builtin(x) 0
#endif
#if __has_builtin(__builtin_amdgcn_global_load_async_to_lds_b128) && \
    __has_builtin(__builtin_amdgcn_s_wait_asynccnt)
#define HAVE_ASYNC 1
#else
#define HAVE_ASYNC 0
#endif

typedef __attribute__((__vector_size__(16))) int i32x4;

typedef __attribute__((ext_vector_type(16))) __bf16 v16bf;
typedef __attribute__((ext_vector_type(8)))  float  v8f;

union Frag { v16bf bf; uint32_t u[8]; };

__device__ __forceinline__ uint16_t f2bf(float f) {
    uint32_t u = __float_as_uint(f);
    u += 0x7FFFu + ((u >> 16) & 1u);     // round-to-nearest-even
    return (uint16_t)(u >> 16);
}

__device__ __forceinline__ v8f wmma_bf16(const Frag& a, const Frag& b, v8f c) {
    return __builtin_amdgcn_wmma_f32_16x16x32_bf16(false, a.bf, false, b.bf,
                                                   (short)0, c, false, false);
}

// 16-byte global -> LDS copy; CDNA5 async DMA when available, sync fallback otherwise.
__device__ __forceinline__ void cp16(const uint16_t* g, uint16_t* l) {
#if HAVE_ASYNC
    __builtin_amdgcn_global_load_async_to_lds_b128(
        (__attribute__((address_space(1))) i32x4*)g,
        (__attribute__((address_space(3))) i32x4*)l, 0, 0);
#else
    *(uint4*)l = *(const uint4*)g;
#endif
}
__device__ __forceinline__ void wait_async() {
#if HAVE_ASYNC
    __builtin_amdgcn_s_wait_asynccnt(0);
#endif
}

// ---------------- BatchNorm: two-stage deterministic reduction ----------------
__global__ __launch_bounds__(512) void bn_partial(const float* __restrict__ x,
                                                  float* __restrict__ part) {
    const int c   = threadIdx.x;          // 512 channels
    const int blk = blockIdx.x;           // 64 blocks
    const int rows = M_ / 64;             // 256 rows each
    const float* p = x + (size_t)blk * rows * C_ + c;
    float s = 0.f, s2 = 0.f;
    for (int r = 0; r < rows; ++r) { float v = p[(size_t)r * C_]; s += v; s2 += v * v; }
    part[blk * C_ + c]            = s;
    part[64 * C_ + blk * C_ + c]  = s2;
}

__global__ __launch_bounds__(512) void bn_finalize(const float* __restrict__ part,
                                                   const float* __restrict__ gamma,
                                                   const float* __restrict__ beta,
                                                   float* __restrict__ sb) {
    const int c = threadIdx.x;
    float s = 0.f, s2 = 0.f;
    for (int b = 0; b < 64; ++b) { s += part[b * C_ + c]; s2 += part[64 * C_ + b * C_ + c]; }
    const float mean  = s  / (float)M_;
    const float var   = s2 / (float)M_ - mean * mean;
    const float scale = gamma[c] * rsqrtf(var + EPSV);
    sb[c]       = scale;
    sb[C_ + c]  = beta[c] - mean * scale;
}

__global__ __launch_bounds__(256) void bn_apply(const float* __restrict__ x,
                                                const float* __restrict__ sb,
                                                uint16_t* __restrict__ feat) {
    const int i = blockIdx.x * 256 + threadIdx.x;   // M_*C_ elements
    const int c = i & (C_ - 1);
    const float v = fmaxf(0.f, x[i] * sb[c] + sb[C_ + c]);
    feat[i] = f2bf(v);
}

__global__ __launch_bounds__(256) void cvt_weights(const float* __restrict__ tw,
                                                   const float* __restrict__ pw,
                                                   const float* __restrict__ gw,
                                                   const float* __restrict__ ww,
                                                   uint16_t* __restrict__ twb,
                                                   uint16_t* __restrict__ pwb,
                                                   uint16_t* __restrict__ gwb,
                                                   uint16_t* __restrict__ wwb) {
    const int i = blockIdx.x * 256 + threadIdx.x;   // CI_*C_ = 131072 each
    twb[i] = f2bf(tw[i]);
    pwb[i] = f2bf(pw[i]);
    gwb[i] = f2bf(gw[i]);
    wwb[i] = f2bf(ww[i]);
}

// 32x32 LDS-tiled bf16 transpose: in (M_ x CI_) -> out (CI_ x M_), coalesced both sides.
__global__ __launch_bounds__(256) void transpose_bf16(const uint16_t* __restrict__ in,
                                                      uint16_t* __restrict__ out) {
    __shared__ uint16_t tile[32][33];
    const int rowbase = blockIdx.x * 32;
    const int colbase = blockIdx.y * 32;
    const int tx = threadIdx.x, ty = threadIdx.y;   // block (32,8)
    for (int j = 0; j < 4; ++j)
        tile[ty + j * 8][tx] = in[(size_t)(rowbase + ty + j * 8) * CI_ + colbase + tx];
    __syncthreads();
    for (int j = 0; j < 4; ++j)
        out[(size_t)(colbase + ty + j * 8) * M_ + rowbase + tx] = tile[tx][ty + j * 8];
}

// ---------------- Generic WMMA GEMM: out[m,n] = A[m,:K] . Bw[n,:K] + bias[n] ----------------
// Double-buffered LDS staging (async DMA when available), 128x64 workgroup tile,
// per wave 32x32 via 2x2 v_wmma_f32_16x16x32_bf16.
template<int K, bool RES>
__global__ __launch_bounds__(256) void gemm_wmma(const uint16_t* __restrict__ A,
                                                 const uint16_t* __restrict__ Bw,
                                                 const float* __restrict__ bias,
                                                 const float* __restrict__ resid,
                                                 uint16_t* __restrict__ outBf,
                                                 float* __restrict__ outF,
                                                 int Nout) {
    __shared__ uint16_t As[2][128 * 32];
    __shared__ uint16_t Bs[2][64 * 32];

    const int tid  = threadIdx.x;
    const int lane = tid & 31, wave = tid >> 5;
    const int wm = wave & 3, wn = wave >> 2;
    const int half = lane >> 4, l15 = lane & 15;
    const int rowbase = blockIdx.x * 128;
    const int nbase   = blockIdx.y * 64;
    constexpr int KC = K / 32;

    auto stage = [&](int kc, int buf) {
        // A tile 128x32 bf16 = 512 x 16B units, 2 per thread
        for (int i = 0; i < 2; ++i) {
            const int u = tid + i * 256;
            cp16(A + (size_t)(rowbase + (u >> 2)) * K + kc * 32 + (u & 3) * 8,
                 As[buf] + u * 8);
        }
        // B tile 64x32 bf16 = 256 x 16B units, 1 per thread
        cp16(Bw + (size_t)(nbase + (tid >> 2)) * K + kc * 32 + (tid & 3) * 8,
             Bs[buf] + tid * 8);
    };

    v8f acc[2][2];
    for (int i = 0; i < 2; ++i)
        for (int j = 0; j < 2; ++j)
            for (int e = 0; e < 8; ++e) acc[i][j][e] = 0.f;

    stage(0, 0);
    wait_async();
    __syncthreads();

    int cur = 0;
    for (int kc = 0; kc < KC; ++kc) {
        if (kc + 1 < KC) stage(kc + 1, cur ^ 1);

        const uint32_t* AsU = (const uint32_t*)As[cur];
        const uint32_t* BsU = (const uint32_t*)Bs[cur];
        Frag af[2], bfr[2];
        for (int mi = 0; mi < 2; ++mi) {       // A frag: lane = row M, K halves by lane-half
            const int m = wm * 32 + mi * 16 + l15;
            const int base = m * 16 + half * 4;
            for (int j = 0; j < 4; ++j) { af[mi].u[j] = AsU[base + j]; af[mi].u[4 + j] = AsU[base + 8 + j]; }
        }
        for (int ni = 0; ni < 2; ++ni) {       // B frag: lane = col N, K halves by lane-half
            const int n = wn * 32 + ni * 16 + l15;
            const int base = n * 16 + half * 4;
            for (int j = 0; j < 4; ++j) { bfr[ni].u[j] = BsU[base + j]; bfr[ni].u[4 + j] = BsU[base + 8 + j]; }
        }
        for (int mi = 0; mi < 2; ++mi)
            for (int ni = 0; ni < 2; ++ni)
                acc[mi][ni] = wmma_bf16(af[mi], bfr[ni], acc[mi][ni]);

        wait_async();
        __syncthreads();
        cur ^= 1;
    }

    for (int mi = 0; mi < 2; ++mi)
        for (int ni = 0; ni < 2; ++ni) {
            const int col = nbase + wn * 32 + ni * 16 + l15;
            const float bb = bias[col];
            for (int v = 0; v < 8; ++v) {
                const int row = rowbase + wm * 32 + mi * 16 + half * 8 + v;
                const size_t idx = (size_t)row * Nout + col;
                const float val = acc[mi][ni][v] + bb;
                if (RES) outF[idx] = resid[idx] + val;
                else     outBf[idx] = f2bf(val);
            }
        }
}

// ---------------- Flash attention ----------------
// Per wave: 16 q-rows x d=256 accumulators. 32-key chunks, double-buffered async
// K/V staging (V pre-transposed in global as gT[d][row]), pipelined fragment loads.
__global__ __launch_bounds__(256) void attn_flash(const uint16_t* __restrict__ T,
                                                  const uint16_t* __restrict__ P,
                                                  const uint16_t* __restrict__ gT,
                                                  uint16_t* __restrict__ Y) {
    __shared__ uint16_t Qs[128 * 256];        // 64 KB
    __shared__ uint16_t Ks[2][32 * 256];      // 2 x 16 KB, [key][c]
    __shared__ uint16_t Vt[2][256 * 32];      // 2 x 16 KB, [d][key]
    __shared__ uint16_t Pscr[8][16 * 32];     //  8 KB per-wave C->A transpose scratch

    const int tid = threadIdx.x, lane = tid & 31, wave = tid >> 5;
    const int half = lane >> 4, l15 = lane & 15;
    const int b = blockIdx.y;
    const size_t qbase = (size_t)b * N_ + blockIdx.x * 128;

    auto stage_kv = [&](int kt, int buf) {
        const uint16_t* ksrc = P + ((size_t)b * N_ + kt * 32) * CI_;  // 32x256 contiguous
        for (int i = 0; i < 4; ++i) {
            const int u = tid + i * 256;                               // 1024 x 16B units
            cp16(ksrc + u * 8, Ks[buf] + u * 8);
        }
        for (int i = 0; i < 4; ++i) {
            const int u = tid + i * 256;                               // d = u>>2, q = u&3
            cp16(gT + (size_t)(u >> 2) * M_ + (size_t)b * N_ + kt * 32 + (u & 3) * 8,
                 Vt[buf] + u * 8);
        }
    };

    {   // stage Q tile (128x256 bf16 = 4096 x 16B units, 16 per thread)
        const uint16_t* qsrc = T + qbase * CI_;
        for (int i = 0; i < 16; ++i) {
            const int u = tid + i * 256;
            cp16(qsrc + u * 8, Qs + u * 8);
        }
    }
    stage_kv(0, 0);

    v8f yacc[16];
    for (int t = 0; t < 16; ++t)
        for (int e = 0; e < 8; ++e) yacc[t][e] = 0.f;
    float mrow[8], lrow[8];
    for (int v = 0; v < 8; ++v) { mrow[v] = -3.0e38f; lrow[v] = 0.f; }

    const uint32_t* QsU = (const uint32_t*)Qs;

    wait_async();
    __syncthreads();

    int cur = 0;
    for (int kt = 0; kt < N_ / 32; ++kt) {
        if (kt + 1 < N_ / 32) stage_kv(kt + 1, cur ^ 1);   // async DMA overlaps compute

        const uint32_t* KsU = (const uint32_t*)Ks[cur];
        const uint32_t* VtU = (const uint32_t*)Vt[cur];

        // scores: S(16q x 32k), K-dim 256 in 8 steps; pipelined fragment loads
        v8f s0, s1;
        for (int e = 0; e < 8; ++e) { s0[e] = 0.f; s1[e] = 0.f; }
        Frag qf[2], k0[2], k1[2];
        auto load_score = [&](int kk, int idx) {
            const int qrow = wave * 16 + l15;
            const int qb = qrow * 128 + kk * 16 + half * 4;
            for (int j = 0; j < 4; ++j) { qf[idx].u[j] = QsU[qb + j]; qf[idx].u[4 + j] = QsU[qb + 8 + j]; }
            const int kb = kk * 16 + half * 4;
            const uint32_t* kr0 = KsU + l15 * 128;
            const uint32_t* kr1 = KsU + (16 + l15) * 128;
            for (int j = 0; j < 4; ++j) { k0[idx].u[j] = kr0[kb + j]; k1[idx].u[j] = kr1[kb + j]; k0[idx].u[4 + j] = kr0[kb + 8 + j]; k1[idx].u[4 + j] = kr1[kb + 8 + j]; }
        };
        load_score(0, 0);
        for (int kk = 0; kk < 8; ++kk) {
            const int ci = kk & 1;
            if (kk < 7) load_score(kk + 1, ci ^ 1);
            s0 = wmma_bf16(qf[ci], k0[ci], s0);
            s1 = wmma_bf16(qf[ci], k1[ci], s1);
        }

        // online softmax (row v + 8*half), shfl reductions within 16-lane halves
        float alpha[8];
        for (int v = 0; v < 8; ++v) {
            float mx = fmaxf(s0[v], s1[v]);
            mx = fmaxf(mx, __shfl_xor(mx, 1, 32));
            mx = fmaxf(mx, __shfl_xor(mx, 2, 32));
            mx = fmaxf(mx, __shfl_xor(mx, 4, 32));
            mx = fmaxf(mx, __shfl_xor(mx, 8, 32));
            const float mnew = fmaxf(mrow[v], mx);
            const float a  = __expf(mrow[v] - mnew);
            const float p0 = __expf(s0[v] - mnew);
            const float p1 = __expf(s1[v] - mnew);
            s0[v] = p0; s1[v] = p1;
            float rs = p0 + p1;
            rs += __shfl_xor(rs, 1, 32);
            rs += __shfl_xor(rs, 2, 32);
            rs += __shfl_xor(rs, 4, 32);
            rs += __shfl_xor(rs, 8, 32);
            lrow[v] = lrow[v] * a + rs;
            mrow[v] = mnew;
            alpha[v] = a;
        }
        for (int t = 0; t < 16; ++t)
            for (int v = 0; v < 8; ++v) yacc[t][v] *= alpha[v];

        // C-layout -> A-layout transpose of P via wave-private LDS (in-order DS per wave)
        uint16_t* ps = Pscr[wave];
        for (int v = 0; v < 8; ++v) {
            const int r = half * 8 + v;
            ps[r * 32 + l15]      = f2bf(s0[v]);
            ps[r * 32 + 16 + l15] = f2bf(s1[v]);
        }
        Frag pa;
        {
            const uint32_t* pr = (const uint32_t*)ps + l15 * 16;
            const int base = half * 4;
            for (int j = 0; j < 4; ++j) { pa.u[j] = pr[base + j]; pa.u[4 + j] = pr[base + 8 + j]; }
        }

        // Y(16x256) += P(16x32) . V(32x256); pipelined V fragment loads
        Frag vb[2];
        auto load_v = [&](int dt, int idx) {
            const uint32_t* vr = VtU + (dt * 16 + l15) * 16;
            const int base = half * 4;
            for (int j = 0; j < 4; ++j) { vb[idx].u[j] = vr[base + j]; vb[idx].u[4 + j] = vr[base + 8 + j]; }
        };
        load_v(0, 0);
        for (int dt = 0; dt < 16; ++dt) {
            const int ci = dt & 1;
            if (dt < 15) load_v(dt + 1, ci ^ 1);
            yacc[dt] = wmma_bf16(pa, vb[ci], yacc[dt]);
        }

        wait_async();
        __syncthreads();
        cur ^= 1;
    }

    // finalize: divide by row sum, store bf16
    for (int v = 0; v < 8; ++v) lrow[v] = 1.0f / lrow[v];
    for (int dt = 0; dt < 16; ++dt) {
        const int col = dt * 16 + l15;
        for (int v = 0; v < 8; ++v) {
            const size_t row = qbase + wave * 16 + half * 8 + v;
            Y[row * CI_ + col] = f2bf(yacc[dt][v] * lrow[v]);
        }
    }
}

// ---------------- host-side orchestration ----------------
extern "C" void kernel_launch(void* const* d_in, const int* in_sizes, int n_in,
                              void* d_out, int out_size, void* d_ws, size_t ws_size,
                              hipStream_t stream) {
    const float* x     = (const float*)d_in[0];
    const float* gamma = (const float*)d_in[1];
    const float* beta  = (const float*)d_in[2];
    const float* tw    = (const float*)d_in[3];
    const float* tb    = (const float*)d_in[4];
    const float* pw    = (const float*)d_in[5];
    const float* pb    = (const float*)d_in[6];
    const float* gw    = (const float*)d_in[7];
    const float* gb    = (const float*)d_in[8];
    const float* ww    = (const float*)d_in[9];
    const float* wb    = (const float*)d_in[10];

    char* ws = (char*)d_ws;
    size_t off = 0;
    auto alloc = [&](size_t bytes) -> char* {
        char* p = ws + off;
        off = (off + bytes + 255) & ~(size_t)255;
        return p;
    };
    float*    part = (float*)   alloc((size_t)2 * 64 * C_ * sizeof(float));
    float*    sb   = (float*)   alloc((size_t)2 * C_ * sizeof(float));
    uint16_t* feat = (uint16_t*)alloc((size_t)M_ * C_  * 2);
    uint16_t* tbf  = (uint16_t*)alloc((size_t)M_ * CI_ * 2);
    uint16_t* pbf  = (uint16_t*)alloc((size_t)M_ * CI_ * 2);
    uint16_t* gbf  = (uint16_t*)alloc((size_t)M_ * CI_ * 2);
    uint16_t* gT   = (uint16_t*)alloc((size_t)CI_ * M_ * 2);
    uint16_t* ybf  = (uint16_t*)alloc((size_t)M_ * CI_ * 2);
    uint16_t* twb  = (uint16_t*)alloc((size_t)CI_ * C_ * 2);
    uint16_t* pwb  = (uint16_t*)alloc((size_t)CI_ * C_ * 2);
    uint16_t* gwb  = (uint16_t*)alloc((size_t)CI_ * C_ * 2);
    uint16_t* wwb  = (uint16_t*)alloc((size_t)C_ * CI_ * 2);

    bn_partial <<<64, 512, 0, stream>>>(x, part);
    bn_finalize<<<1, 512, 0, stream>>>(part, gamma, beta, sb);
    bn_apply   <<<(M_ * C_) / 256, 256, 0, stream>>>(x, sb, feat);
    cvt_weights<<<(CI_ * C_) / 256, 256, 0, stream>>>(tw, pw, gw, ww, twb, pwb, gwb, wwb);

    dim3 g1(M_ / 128, CI_ / 64);   // (128, 4)
    gemm_wmma<C_, false><<<g1, 256, 0, stream>>>(feat, twb, tb, nullptr, tbf, nullptr, CI_);
    gemm_wmma<C_, false><<<g1, 256, 0, stream>>>(feat, pwb, pb, nullptr, pbf, nullptr, CI_);
    gemm_wmma<C_, false><<<g1, 256, 0, stream>>>(feat, gwb, gb, nullptr, gbf, nullptr, CI_);

    transpose_bf16<<<dim3(M_ / 32, CI_ / 32), dim3(32, 8), 0, stream>>>(gbf, gT);

    attn_flash<<<dim3(N_ / 128, B_), 256, 0, stream>>>(tbf, pbf, gT, ybf);

    gemm_wmma<CI_, true><<<dim3(M_ / 128, C_ / 64), 256, 0, stream>>>(
        ybf, wwb, wb, x, nullptr, (float*)d_out, C_);
}
